// SocialPooling_28381143892376
// MI455X (gfx1250) — matrix operand order repaired and checked by reading, not compiled, not run on hardware
//
#include <hip/hip_runtime.h>
#include <hip/hip_bf16.h>
#include <stdint.h>

// ---------------------------------------------------------------------------
// SocialPooling on MI455X (gfx1250, wave32)
//
//   pooled[i,g,h] = sum_j W_g[i,j] * ht[j,h],  W_g in {0,1}
//
//   * 1 workgroup = 128 threads = 4 wave32; workgroup owns 16 output rows.
//   * wave g owns grid-cell g; 4x (16x16 f32) accumulators cover H=64.
//   * scene ids are SORTED -> binary search bounds the j range per i-block.
//   * ht f32 split hi/lo bf16 ONCE by a pre-pass into d_ws, already swizzled
//     into WMMA B-operand order -> hot loop B build = 2x ds_load_b128.
//   * 8KB j-chunk staged to LDS by the Tensor Data Mover
//     (tensor_load_to_lds + s_wait_tensorcnt), fallback = cooperative loads.
//   * 2x v_wmma_f32_16x16x32_bf16 per (g, h-block) per k-step (hi + lo).
//   * USE_PRE is a template parameter so the hot loop has no runtime branch.
// ---------------------------------------------------------------------------

typedef __attribute__((ext_vector_type(16))) __bf16          v16bf;
typedef __attribute__((ext_vector_type(16))) unsigned short  v16u;
typedef __attribute__((ext_vector_type(8)))  float           v8f;
typedef __attribute__((ext_vector_type(4)))  unsigned int    u32x4;
typedef __attribute__((ext_vector_type(8)))  int             i32x8;
typedef __attribute__((ext_vector_type(4)))  int             i32x4;

struct U32x8 { uint4 a, b; };   // 32B container for bit_cast to v16bf

#define NPTS   4096
#define HID    64
#define NG     4
#define JTILE  32
#define HALF_F 0.8f
#define CELL_F 0.8f
#define EPS_F  0.01f
// pre-swizzled ws: per 32-row chunk: [hilo][hcol(64)][khalf(2)][e(16)] u16 = 8KB
#define CHUNK_U16 4096
#define WS_NEED   ((size_t)(NPTS / JTILE) * CHUNK_U16 * sizeof(unsigned short))

// ---- 8KB contiguous global->LDS copy via Tensor Data Mover ----------------
__device__ __forceinline__ void tdm_copy8k(const void* src, void* lds_dst) {
#if __has_builtin(__builtin_amdgcn_tensor_load_to_lds)
    const unsigned long long ga = (unsigned long long)src;
    const unsigned int ldsb = (unsigned int)(unsigned long long)lds_dst;
    // D# (CDNA5 ISA): 1 row of 2048 dwords, data_size=4B.
    u32x4 d0 = { 1u,                                    // count=1, user D#
                 ldsb,                                  // lds_addr
                 (unsigned int)ga,                      // global_addr[31:0]
                 ((unsigned int)(ga >> 32) & 0x01FFFFFFu) | (2u << 30) };
    i32x8 d1 = { (int)(2u << 16),                      // data_size=2 -> 4 bytes
                 (int)(2048u << 16),                   // tensor_dim0 = 2048 (lo16)
                 (int)(1u << 16),                      // tdim0 hi=0 | tensor_dim1=1
                 (int)(2048u << 16),                   // tdim1 hi=0 | tile_dim0=2048
                 1,                                    // tile_dim1=1 | tile_dim2=0
                 2048,                                 // tensor_dim0_stride lo32
                 0, 0 };
    i32x4 z4 = { 0, 0, 0, 0 };
    i32x8 z8 = { 0, 0, 0, 0, 0, 0, 0, 0 };
    __builtin_amdgcn_tensor_load_to_lds(d0, d1, z4, z4, z8, 0);
    __builtin_amdgcn_s_wait_tensorcnt(0);
#endif
}

// ---- pre-pass: ht f32 -> swizzled bf16 hi/lo in ws ------------------------
__global__ __launch_bounds__(256) void split_ht_kernel(
    const float* __restrict__ ht, unsigned short* __restrict__ ws) {
    const int t = blockIdx.x * 256 + threadIdx.x;      // 0 .. N*HID-1
    if (t >= NPTS * HID) return;
    const int e  = t & 15;
    const int kh = (t >> 4) & 1;
    const int h  = (t >> 5) & 63;
    const int c  = t >> 11;                            // 32-row chunk
    const int j  = c * JTILE + kh * 16 + e;
    const float f = ht[(size_t)j * HID + h];
    const unsigned int  u  = __builtin_bit_cast(unsigned int, f);
    const unsigned short hh = (unsigned short)(u >> 16);            // bf16 hi
    const float fh = __builtin_bit_cast(float, (unsigned int)hh << 16);
    const float fl = f - fh;
    const unsigned short ll =
        (unsigned short)(__builtin_bit_cast(unsigned int, fl) >> 16); // bf16 lo
    const size_t base = (size_t)c * CHUNK_U16 + ((size_t)h * 2 + kh) * 16 + e;
    ws[base]        = hh;                              // hilo=0
    ws[base + 2048] = ll;                              // hilo=1
}

template <int USE_PRE>
__global__ __launch_bounds__(128) void social_pool_wmma(
    const float*          __restrict__ ht,     // [N,1,64] f32
    const float*          __restrict__ pos,    // [N,1,2]  f32
    const long long*      __restrict__ scene,  // [N,1]    i64 (sorted)
    float*                __restrict__ out,    // [N,4,64] f32
    const unsigned short* __restrict__ ws)     // pre-split bf16 (or unused)
{
    __shared__ __align__(32) unsigned char lds_buf[JTILE * HID * 4];  // 8 KB
    __shared__ float lds_px[JTILE];
    __shared__ float lds_py[JTILE];
    __shared__ int   lds_sc[JTILE];
    float* const lds_ht = (float*)lds_buf;             // f32 path view
    const uint4* const lds_q = (const uint4*)lds_buf;  // bf16 path view (16B)

    const int tid  = threadIdx.x;
    const int lane = tid & 31;
    const int g    = tid >> 5;        // wave id == grid cell
    const int i0   = blockIdx.x * 16;
    const int m    = lane & 15;       // row-in-tile (A) / column (B,C)
    const int hi16 = lane >> 4;       // lane half select

    const int   irow = i0 + m;
    const float pix  = pos[irow * 2 + 0];
    const float piy  = pos[irow * 2 + 1];
    const int   isc  = (int)scene[irow];

    // ---- uniform: bound j by scene range (scene sorted ascending) ----------
    const long long smin = scene[i0];
    const long long smax = scene[i0 + 15];
    int lo = 0, hi = NPTS;
    while (lo < hi) { int mid = (lo + hi) >> 1; if (scene[mid] <  smin) lo = mid + 1; else hi = mid; }
    const int jlo = lo & ~(JTILE - 1);
    lo = 0; hi = NPTS;
    while (lo < hi) { int mid = (lo + hi) >> 1; if (scene[mid] <= smax) lo = mid + 1; else hi = mid; }
    const int jhi = lo;

    v8f acc[4];
#pragma unroll
    for (int b = 0; b < 4; ++b) acc[b] = (v8f){0.f,0.f,0.f,0.f,0.f,0.f,0.f,0.f};

    for (int j0 = jlo; j0 < jhi; j0 += JTILE) {
        __syncthreads();              // protect LDS from previous iteration

        if (g == 0) {
            const int j = j0 + lane;
            lds_px[lane] = pos[j * 2 + 0];
            lds_py[lane] = pos[j * 2 + 1];
            lds_sc[lane] = (int)scene[j];
        }

        // stage 8KB: pre-swizzled bf16 chunk, or raw f32 tile
        const void* src = USE_PRE
            ? (const void*)(ws + (size_t)(j0 >> 5) * CHUNK_U16)
            : (const void*)(ht + (size_t)j0 * HID);
#if __has_builtin(__builtin_amdgcn_tensor_load_to_lds)
        if (g == 0) tdm_copy8k(src, lds_buf);
#else
        {
            const uint4* s = (const uint4*)src;
            uint4*       d = (uint4*)lds_buf;
            for (int t = tid; t < 512; t += 128) d[t] = s[t];
        }
#endif
        __syncthreads();

        // ---- A_g tile: 16x32 bf16 weights, contiguous k order (k = 16*half + e) -------------
        // lane<16 : K in {0..15}; lane>=16 : K in {16..31}
        v16u aw = {0,0,0,0,0,0,0,0,0,0,0,0,0,0,0,0};
#pragma unroll
        for (int e = 0; e < 16; ++e) {
            const int k = e + (hi16 ? 16 : 0);
            const int j = j0 + k;
            const float rx = lds_px[k] - pix;
            const float ry = lds_py[k] - piy;
            const bool within = (rx <  (HALF_F - EPS_F)) && (rx > -(HALF_F - EPS_F)) &&
                                (ry <  (HALF_F - EPS_F)) && (ry > -(HALF_F - EPS_F));
            const bool same   = (lds_sc[k] == isc);
            const bool nself  = (j != irow);
            const int  gx  = (int)floorf((rx + HALF_F) / CELL_F);
            const int  gy  = (int)floorf((ry + HALF_F) / CELL_F);
            const int  gid = gx * 2 + gy;
            aw[e] = (within && same && nself && (gid == g))
                        ? (unsigned short)0x3F80 /* bf16 1.0 */ : (unsigned short)0;
        }
        const v16bf a = __builtin_bit_cast(v16bf, aw);

        // ---- B tiles (32x16 bf16 hi/lo) + WMMA, per 16-wide h block --------
        // B layout: lane<16 K=0..15, lane>=16 K=16..31, element e == K offset.
#pragma unroll
        for (int b = 0; b < 4; ++b) {
            const int hcol = b * 16 + m;
            v16bf bh, bl;
            if (USE_PRE) {
                // swizzled LDS, uint4 index = ((hilo*64+hcol)*2 + khalf)*2 + q
                const int qi = (hcol * 2 + hi16) * 2;
                U32x8 ph = { lds_q[qi],       lds_q[qi + 1] };       // 2x b128
                U32x8 pl = { lds_q[qi + 256], lds_q[qi + 257] };     // +8KB/2
                bh = __builtin_bit_cast(v16bf, ph);
                bl = __builtin_bit_cast(v16bf, pl);
            } else {
                v16u bh_ = {0,0,0,0,0,0,0,0,0,0,0,0,0,0,0,0};
                v16u bl_ = {0,0,0,0,0,0,0,0,0,0,0,0,0,0,0,0};
#pragma unroll
                for (int e = 0; e < 16; ++e) {
                    const int k = e + (hi16 ? 16 : 0);
                    const float f = lds_ht[k * HID + hcol];
                    const unsigned int u  = __builtin_bit_cast(unsigned int, f);
                    const unsigned short hh = (unsigned short)(u >> 16);
                    const float fh = __builtin_bit_cast(float, (unsigned int)hh << 16);
                    const float fl = f - fh;
                    bh_[e] = hh;
                    bl_[e] = (unsigned short)(__builtin_bit_cast(unsigned int, fl) >> 16);
                }
                bh = __builtin_bit_cast(v16bf, bh_);
                bl = __builtin_bit_cast(v16bf, bl_);
            }
            acc[b] = __builtin_amdgcn_wmma_f32_16x16x32_bf16(
                         false, a, false, bh, (short)0, acc[b], false, false);
            acc[b] = __builtin_amdgcn_wmma_f32_16x16x32_bf16(
                         false, a, false, bl, (short)0, acc[b], false, false);
        }
    }

    // ---- epilogue: C layout VGPR r -> row r (lane<16) / r+8 (lane>=16) -----
#pragma unroll
    for (int b = 0; b < 4; ++b) {
#pragma unroll
        for (int r = 0; r < 8; ++r) {
            const int i = i0 + r + (hi16 ? 8 : 0);
            out[((size_t)i * NG + g) * HID + b * 16 + m] = acc[b][r];
        }
    }
}

extern "C" void kernel_launch(void* const* d_in, const int* in_sizes, int n_in,
                              void* d_out, int out_size, void* d_ws, size_t ws_size,
                              hipStream_t stream) {
    (void)in_sizes; (void)n_in; (void)out_size;
    const float*     ht    = (const float*)d_in[0];
    const float*     pos   = (const float*)d_in[1];
    const long long* scene = (const long long*)d_in[2];
    float*           out   = (float*)d_out;

    const dim3 grid(NPTS / 16), block(128);
    if (d_ws != nullptr && ws_size >= WS_NEED) {
        split_ht_kernel<<<dim3((NPTS * HID) / 256), dim3(256), 0, stream>>>(
            ht, (unsigned short*)d_ws);
        social_pool_wmma<1><<<grid, block, 0, stream>>>(
            ht, pos, scene, out, (const unsigned short*)d_ws);
    } else {
        social_pool_wmma<0><<<grid, block, 0, stream>>>(
            ht, pos, scene, out, nullptr);
    }
}
